// MambaMixer_6631429505697
// MI455X (gfx1250) — matrix operand order repaired and checked by reading, not compiled, not run on hardware
//
#include <hip/hip_runtime.h>
#include <math.h>

// ---------------------------------------------------------------------------
// Mamba mixer forward for MI455X (gfx1250, wave32, WMMA).
// GEMMs use bf16x3 split-precision via v_wmma_f32_16x16x32_bf16.
// Big GEMMs are register-blocked 32x64 per wave (8 accumulators) to lift
// wave-level arithmetic intensity from ~12 to ~33 FLOP/B.
// ---------------------------------------------------------------------------

typedef __attribute__((ext_vector_type(16))) __bf16 v16bf;
typedef __attribute__((ext_vector_type(8)))  __bf16 bf16x8;
typedef __attribute__((ext_vector_type(8)))  float  v8f;

#define L_SEQ   2048
#define DIM_    1024
#define DI_     2048   // d_inner
#define NXZ_    4096   // 2*d_inner
#define RNK_    64     // dt_rank
#define NDB_    96     // dt_rank + 2*d_state
#define NST_    16     // d_state

#define WMMA_BF16(A, B, Cacc) \
    __builtin_amdgcn_wmma_f32_16x16x32_bf16(false, (A), false, (B), (short)0, (Cacc), false, false)

// ---------------------------------------------------------------------------
// Split an fp32 array into hi/lo bf16 pair:  v ~= (float)hi + (float)lo
// ---------------------------------------------------------------------------
__global__ __launch_bounds__(256)
void split_bf16_kernel(const float* __restrict__ src,
                       __bf16* __restrict__ hi, __bf16* __restrict__ lo,
                       int count) {
    int i = blockIdx.x * blockDim.x + threadIdx.x;
    if (i >= count) return;
    float v = src[i];
    __bf16 h = (__bf16)v;
    hi[i] = h;
    lo[i] = (__bf16)(v - (float)h);
}

// ---------------------------------------------------------------------------
// Build a 16-element bf16 fragment from two contiguous 8-element runs.
// ---------------------------------------------------------------------------
__device__ __forceinline__ v16bf frag16(const __bf16* p0, const __bf16* p1) {
    bf16x8 a = *(const bf16x8*)p0;
    bf16x8 b = *(const bf16x8*)p1;
    return __builtin_shufflevector(a, b, 0,1,2,3,4,5,6,7,8,9,10,11,12,13,14,15);
}

__device__ __forceinline__ float softplus_ep(float v, float b2) {
    v += b2;
    return (v > 20.0f) ? v : log1pf(__expf(v));
}

// ---------------------------------------------------------------------------
// Register-blocked GEMM: each wave computes a 32x64 output macro-tile
// (2 M-subtiles x 4 N-subtiles). C[m,n] = sum_k A[m,k] * W[n,k].
// Requires M%32==0, N%64==0, K%32==0.
// mode 0: plain store; mode 1: softplus(acc + 2*bias[n]) (dt path).
// ---------------------------------------------------------------------------
__global__ __launch_bounds__(256)
void wmma_gemm_bf16x3_blk(const __bf16* __restrict__ Ahi, const __bf16* __restrict__ Alo, int lda,
                          const __bf16* __restrict__ Whi, const __bf16* __restrict__ Wlo, int ldw,
                          float* __restrict__ C, int ldc,
                          int M, int N, int K,
                          int mode, const float* __restrict__ bias) {
    const int wave   = threadIdx.x >> 5;
    const int lane   = threadIdx.x & 31;
    const int tilesN = N >> 6;                              // 64-col macro tiles
    const int tile   = blockIdx.x * (blockDim.x >> 5) + wave;  // wave-uniform
    if (tile >= (M >> 5) * tilesN) return;                  // whole-wave exit
    const int mt = tile / tilesN;
    const int nt = tile % tilesN;
    const int r  = lane & 15;
    const int h  = lane >> 4;

    const __bf16 *aH[2], *aL[2], *wH[4], *wL[4];
#pragma unroll
    for (int mi = 0; mi < 2; ++mi) {
        aH[mi] = Ahi + (size_t)(mt * 32 + mi * 16 + r) * lda;
        aL[mi] = Alo + (size_t)(mt * 32 + mi * 16 + r) * lda;
    }
#pragma unroll
    for (int ni = 0; ni < 4; ++ni) {
        wH[ni] = Whi + (size_t)(nt * 64 + ni * 16 + r) * ldw;
        wL[ni] = Wlo + (size_t)(nt * 64 + ni * 16 + r) * ldw;
    }

    v8f acc[2][4] = {};
    for (int kb = 0; kb < K; kb += 32) {
        v16bf ahi[2], alo[2], bhi[4], blo[4];
#pragma unroll
        for (int mi = 0; mi < 2; ++mi) {
            // A frag: elems 0..7 <- K=kb+8h.., elems 8..15 <- K=kb+16+8h..
            ahi[mi] = frag16(aH[mi] + kb + 8 * h, aH[mi] + kb + 16 + 8 * h);
            alo[mi] = frag16(aL[mi] + kb + 8 * h, aL[mi] + kb + 16 + 8 * h);
        }
#pragma unroll
        for (int ni = 0; ni < 4; ++ni) {
            // B frag: elems 0..7 <- K=kb+8h.., elems 8..15 <- K=kb+16+8h..
            bhi[ni] = frag16(wH[ni] + kb + 8 * h, wH[ni] + kb + 16 + 8 * h);
            blo[ni] = frag16(wL[ni] + kb + 8 * h, wL[ni] + kb + 16 + 8 * h);
        }
        // Grouped so consecutive WMMAs share the A operand (operand reuse).
#pragma unroll
        for (int mi = 0; mi < 2; ++mi) {
#pragma unroll
            for (int ni = 0; ni < 4; ++ni) {
                acc[mi][ni] = WMMA_BF16(ahi[mi], bhi[ni], acc[mi][ni]);
                acc[mi][ni] = WMMA_BF16(ahi[mi], blo[ni], acc[mi][ni]);
                acc[mi][ni] = WMMA_BF16(alo[mi], bhi[ni], acc[mi][ni]);
            }
        }
    }

#pragma unroll
    for (int ni = 0; ni < 4; ++ni) {
        const int col = nt * 64 + ni * 16 + r;
        const float b2 = (mode == 1) ? 2.0f * bias[col] : 0.0f;
#pragma unroll
        for (int mi = 0; mi < 2; ++mi) {
#pragma unroll
            for (int rr = 0; rr < 8; ++rr) {
                int row = mt * 32 + mi * 16 + rr + 8 * h;  // C: VGPR rr -> row rr+8h
                float v = acc[mi][ni][rr];
                if (mode == 1) v = softplus_ep(v, b2);
                C[(size_t)row * ldc + col] = v;
            }
        }
    }
}

// ---------------------------------------------------------------------------
// Simple one-tile-per-wave GEMM (used for the small x_proj GEMM, N=96).
// ---------------------------------------------------------------------------
__global__ __launch_bounds__(256)
void wmma_gemm_bf16x3(const __bf16* __restrict__ Ahi, const __bf16* __restrict__ Alo, int lda,
                      const __bf16* __restrict__ Whi, const __bf16* __restrict__ Wlo, int ldw,
                      float* __restrict__ C, int ldc,
                      int M, int N, int K) {
    const int wave   = threadIdx.x >> 5;
    const int lane   = threadIdx.x & 31;
    const int tilesN = N >> 4;
    const int tile   = blockIdx.x * (blockDim.x >> 5) + wave;
    if (tile >= (M >> 4) * tilesN) return;
    const int mt = tile / tilesN;
    const int nt = tile % tilesN;
    const int r  = lane & 15;
    const int h  = lane >> 4;

    const __bf16* aHi = Ahi + (size_t)(mt * 16 + r) * lda;
    const __bf16* aLo = Alo + (size_t)(mt * 16 + r) * lda;
    const __bf16* wHi = Whi + (size_t)(nt * 16 + r) * ldw;
    const __bf16* wLo = Wlo + (size_t)(nt * 16 + r) * ldw;

    v8f acc = {};
    for (int kb = 0; kb < K; kb += 32) {
        v16bf ahi = frag16(aHi + kb + 8 * h, aHi + kb + 16 + 8 * h);
        v16bf alo = frag16(aLo + kb + 8 * h, aLo + kb + 16 + 8 * h);
        v16bf bhi = frag16(wHi + kb + 8 * h, wHi + kb + 16 + 8 * h);
        v16bf blo = frag16(wLo + kb + 8 * h, wLo + kb + 16 + 8 * h);
        acc = WMMA_BF16(ahi, bhi, acc);
        acc = WMMA_BF16(ahi, blo, acc);
        acc = WMMA_BF16(alo, bhi, acc);
    }
    const int col = nt * 16 + r;
#pragma unroll
    for (int rr = 0; rr < 8; ++rr) {
        int row = mt * 16 + rr + 8 * h;
        C[(size_t)row * ldc + col] = acc[rr];
    }
}

// ---------------------------------------------------------------------------
// Causal depthwise conv (K=4) + bias + SiLU. Reads x = xz[:, 0:DI_].
// Writes u (fp32 for the scan) and u hi/lo bf16 (for the x_proj GEMM).
// ---------------------------------------------------------------------------
__global__ __launch_bounds__(256)
void conv_silu_kernel(const float* __restrict__ xz,
                      const float* __restrict__ conv_w,
                      const float* __restrict__ conv_b,
                      float* __restrict__ u,
                      __bf16* __restrict__ u_hi, __bf16* __restrict__ u_lo) {
    int idx = blockIdx.x * blockDim.x + threadIdx.x;    // l*DI_ + d
    if (idx >= L_SEQ * DI_) return;
    int d = idx & (DI_ - 1);
    int l = idx >> 11;
    float acc = conv_b[d];
#pragma unroll
    for (int j = 0; j < 4; ++j) {
        int ll = l - 3 + j;
        if (ll >= 0) acc += xz[(size_t)ll * NXZ_ + d] * conv_w[d * 4 + j];
    }
    float s = acc / (1.0f + __expf(-acc));             // SiLU
    u[idx] = s;
    __bf16 h = (__bf16)s;
    u_hi[idx] = h;
    u_lo[idx] = (__bf16)(s - (float)h);
}

// ---------------------------------------------------------------------------
// Selective scan. One thread per (channel d, state n); lanes 0..15 / 16..31
// of each wave hold the 16 states of two channels. Sequential over L with a
// 4-hop shfl_xor reduction over n. Fuses +u*D_skip and *silu(z) gating and
// emits y as hi/lo bf16 for the out_proj WMMA GEMM.
// ---------------------------------------------------------------------------
__global__ __launch_bounds__(256)
void scan_kernel(const float* __restrict__ delta,
                 const float* __restrict__ xdb,
                 const float* __restrict__ u,
                 const float* __restrict__ xz,
                 const float* __restrict__ A_log,
                 const float* __restrict__ D_skip,
                 __bf16* __restrict__ y_hi, __bf16* __restrict__ y_lo) {
    int g = blockIdx.x * blockDim.x + threadIdx.x;     // 32768 threads
    int d = g >> 4;
    int n = g & 15;
    float a  = -__expf(A_log[d * NST_ + n]);
    float Dk = D_skip[d];
    float hst = 0.0f;
    for (int l = 0; l < L_SEQ; ++l) {
        float dt = delta[(size_t)l * DI_ + d];
        float ut = u[(size_t)l * DI_ + d];
        const float* row = xdb + (size_t)l * NDB_;
        float bt = row[RNK_ + n];                      // B[l,n]
        float ct = row[RNK_ + NST_ + n];               // C[l,n]
        hst = hst * __expf(dt * a) + (dt * ut) * bt;
        float p = hst * ct;
        // reduce over the 16 states (stays within each 16-lane half)
        p += __shfl_xor(p, 1, 32);
        p += __shfl_xor(p, 2, 32);
        p += __shfl_xor(p, 4, 32);
        p += __shfl_xor(p, 8, 32);
        if (n == 0) {
            float yv = p + ut * Dk;
            float z  = xz[(size_t)l * NXZ_ + DI_ + d];
            yv *= z / (1.0f + __expf(-z));             // * silu(z)
            __bf16 hh = (__bf16)yv;
            y_hi[(size_t)l * DI_ + d] = hh;
            y_lo[(size_t)l * DI_ + d] = (__bf16)(yv - (float)hh);
        }
    }
}

// ---------------------------------------------------------------------------
// Host-side launch
// ---------------------------------------------------------------------------
extern "C" void kernel_launch(void* const* d_in, const int* in_sizes, int n_in,
                              void* d_out, int out_size, void* d_ws, size_t ws_size,
                              hipStream_t stream) {
    const float* hs     = (const float*)d_in[0];   // (L, DIM)
    const float* in_w   = (const float*)d_in[1];   // (4096, 1024)
    const float* conv_w = (const float*)d_in[2];   // (2048, 1, 4)
    const float* conv_b = (const float*)d_in[3];   // (2048,)
    const float* xp_w   = (const float*)d_in[4];   // (96, 2048)
    const float* dt_w   = (const float*)d_in[5];   // (2048, 64)
    const float* dt_b   = (const float*)d_in[6];   // (2048,)
    const float* A_log  = (const float*)d_in[7];   // (2048, 16)
    const float* Dsk    = (const float*)d_in[8];   // (2048,)
    const float* out_w  = (const float*)d_in[9];   // (1024, 2048)

    // ---- carve workspace (256B aligned chunks) ----
    char* base = (char*)d_ws;
    size_t off = 0;
    auto carve = [&](size_t bytes) {
        void* p = base + off;
        off = (off + bytes + 255) & ~(size_t)255;
        return p;
    };
    float*  xz    = (float*) carve((size_t)L_SEQ * NXZ_ * 4);
    float*  u     = (float*) carve((size_t)L_SEQ * DI_  * 4);
    float*  xdb   = (float*) carve((size_t)L_SEQ * NDB_ * 4);
    float*  delta = (float*) carve((size_t)L_SEQ * DI_  * 4);
    __bf16* hs_h  = (__bf16*)carve((size_t)L_SEQ * DIM_ * 2);
    __bf16* hs_l  = (__bf16*)carve((size_t)L_SEQ * DIM_ * 2);
    __bf16* inw_h = (__bf16*)carve((size_t)NXZ_ * DIM_ * 2);
    __bf16* inw_l = (__bf16*)carve((size_t)NXZ_ * DIM_ * 2);
    __bf16* u_h   = (__bf16*)carve((size_t)L_SEQ * DI_ * 2);
    __bf16* u_l   = (__bf16*)carve((size_t)L_SEQ * DI_ * 2);
    __bf16* xpw_h = (__bf16*)carve((size_t)NDB_ * DI_ * 2);
    __bf16* xpw_l = (__bf16*)carve((size_t)NDB_ * DI_ * 2);
    __bf16* xdb_h = (__bf16*)carve((size_t)L_SEQ * NDB_ * 2);
    __bf16* xdb_l = (__bf16*)carve((size_t)L_SEQ * NDB_ * 2);
    __bf16* dtw_h = (__bf16*)carve((size_t)DI_ * RNK_ * 2);
    __bf16* dtw_l = (__bf16*)carve((size_t)DI_ * RNK_ * 2);
    __bf16* y_h   = (__bf16*)carve((size_t)L_SEQ * DI_ * 2);
    __bf16* y_l   = (__bf16*)carve((size_t)L_SEQ * DI_ * 2);
    __bf16* ow_h  = (__bf16*)carve((size_t)DIM_ * DI_ * 2);
    __bf16* ow_l  = (__bf16*)carve((size_t)DIM_ * DI_ * 2);

    const int BLK = 256;
    auto splitN = [&](const float* s, __bf16* h, __bf16* l, int n) {
        split_bf16_kernel<<<(n + BLK - 1) / BLK, BLK, 0, stream>>>(s, h, l, n);
    };
    auto gemm_blk = [&](const __bf16* Ah, const __bf16* Al, int lda,
                        const __bf16* Wh, const __bf16* Wl, int ldw,
                        float* C, int ldc, int M, int N, int K,
                        int mode, const float* bias) {
        int tiles = (M / 32) * (N / 64);
        wmma_gemm_bf16x3_blk<<<(tiles + 7) / 8, BLK, 0, stream>>>(
            Ah, Al, lda, Wh, Wl, ldw, C, ldc, M, N, K, mode, bias);
    };

    // 1) split fp32 operands into bf16 hi/lo pairs (convert once, reuse)
    splitN(hs,    hs_h,  hs_l,  L_SEQ * DIM_);
    splitN(in_w,  inw_h, inw_l, NXZ_ * DIM_);
    splitN(xp_w,  xpw_h, xpw_l, NDB_ * DI_);
    splitN(dt_w,  dtw_h, dtw_l, DI_ * RNK_);
    splitN(out_w, ow_h,  ow_l,  DIM_ * DI_);

    // 2) in_proj: xz = hs @ in_proj_w^T        (2048 x 4096, K=1024)
    gemm_blk(hs_h, hs_l, DIM_, inw_h, inw_l, DIM_, xz, NXZ_,
             L_SEQ, NXZ_, DIM_, 0, nullptr);

    // 3) causal depthwise conv + SiLU -> u (+ bf16 hi/lo)
    conv_silu_kernel<<<(L_SEQ * DI_) / BLK, BLK, 0, stream>>>(
        xz, conv_w, conv_b, u, u_h, u_l);

    // 4) x_proj: xdb = u @ x_proj_w^T          (2048 x 96, K=2048) — small N
    {
        int tiles = (L_SEQ / 16) * (NDB_ / 16);
        wmma_gemm_bf16x3<<<(tiles + 7) / 8, BLK, 0, stream>>>(
            u_h, u_l, DI_, xpw_h, xpw_l, DI_, xdb, NDB_, L_SEQ, NDB_, DI_);
    }
    splitN(xdb, xdb_h, xdb_l, L_SEQ * NDB_);

    // 5) dt_proj + double-bias + softplus: delta (2048 x 2048, K=64)
    gemm_blk(xdb_h, xdb_l, NDB_, dtw_h, dtw_l, RNK_, delta, DI_,
             L_SEQ, DI_, RNK_, 1, dt_b);

    // 6) selective scan + D skip + silu(z) gating -> y (bf16 hi/lo)
    scan_kernel<<<(DI_ * NST_) / BLK, BLK, 0, stream>>>(
        delta, xdb, u, xz, A_log, Dsk, y_h, y_l);

    // 7) out_proj: out = y @ out_proj_w^T      (2048 x 1024, K=2048)
    gemm_blk(y_h, y_l, DI_, ow_h, ow_l, DI_, (float*)d_out, DIM_,
             L_SEQ, DIM_, DI_, 0, nullptr);
}
